// WindowAttention_49194555408561
// MI455X (gfx1250) — matrix operand order repaired and checked by reading, not compiled, not run on hardware
//
#include <hip/hip_runtime.h>
#include <stdint.h>

typedef __bf16 bf16;
typedef __attribute__((ext_vector_type(16))) __bf16 v16bf;
typedef __attribute__((ext_vector_type(8)))  __bf16 v8bf;
typedef __attribute__((ext_vector_type(8)))  float  v8f;
typedef __attribute__((ext_vector_type(4)))  unsigned int v4u;
typedef __attribute__((ext_vector_type(8)))  int v8i_;
typedef __attribute__((ext_vector_type(4)))  int v4i_;

#define HEADS 4
#define HD    32
#define L_    49
#define LP    64
#define NWIN  4096
#define NTOK  200704
#define SCALE 0.17677669529663687f  // hd^-0.5

#if defined(__has_builtin)
#if __has_builtin(__builtin_amdgcn_tensor_load_to_lds)
#define HAVE_TDM 1
#endif
#endif
#ifndef HAVE_TDM
#define HAVE_TDM 0
#endif

// ---- TDM: 1-D contiguous copy global->LDS (D# per ISA 08_async_tensor) ----
#if HAVE_TDM
static __device__ __forceinline__ void tdm_copy_1d(void* lds_ptr, const void* gptr,
                                                   unsigned nelem, unsigned dsz_code) {
  const unsigned lds = (unsigned)(uintptr_t)lds_ptr;        // low 32 bits = LDS offset
  const unsigned long long ga = (unsigned long long)(uintptr_t)gptr;
  // group0: count=1 | lds_addr | global_addr[56:0] | type=2
  v4u g0 = { 1u, lds, (unsigned)ga,
             (unsigned)((ga >> 32) & 0x01FFFFFFull) | 0x80000000u };
  // group1: data_size | tensor_dim0=nelem | tensor_dim1=1 | tile_dim0=nelem |
  //         tile_dim1=1 | stride0=nelem
  v8i_ g1 = { (int)(dsz_code << 16), (int)(nelem << 16), (int)(1u << 16),
              (int)(nelem << 16), 1, (int)nelem, 0, 0 };
  v4i_ z4 = { 0, 0, 0, 0 };
  v8i_ z8 = { 0, 0, 0, 0, 0, 0, 0, 0 };
  __builtin_amdgcn_tensor_load_to_lds(g0, g1, z4, z4, z8, 0);   // 6-arg variant
}
static __device__ __forceinline__ void tdm_wait() {
#if __has_builtin(__builtin_amdgcn_s_wait_tensorcnt)
  __builtin_amdgcn_s_wait_tensorcnt((short)0);
#else
  asm volatile("s_wait_tensorcnt 0x0" ::: "memory");
#endif
}
#endif

// ---- WMMA fragment helpers (CDNA5 ISA 7.12.2, wave32) ---------------------
static __device__ __forceinline__ v16bf load_A16(const bf16* p) {
  v8bf lo = *(const v8bf*)p;
  v8bf hi = *(const v8bf*)(p + 8);
  return __builtin_shufflevector(lo, hi, 0, 1, 2, 3, 4, 5, 6, 7,
                                 8, 9, 10, 11, 12, 13, 14, 15);
}
static __device__ __forceinline__ v8f wmma_bf16(v16bf a, v16bf b, v8f c) {
  return __builtin_amdgcn_wmma_f32_16x16x32_bf16(false, a, false, b, (short)0, c,
                                                 false, false);
}

// ---- init: convert weights to bf16 in B-fragment-swizzled order -----------
// Wsw[((jt*4+kk)*32 + lane)*16 + i] = W[kk*32 + 16*(lane>>4) + i][jt*16 + (lane&15)]
__global__ __launch_bounds__(256) void init_weights_kernel(
    const float* __restrict__ qkv_w, const float* __restrict__ proj_w,
    bf16* __restrict__ Wq, bf16* __restrict__ Wp) {
  const int idx = blockIdx.x * 256 + threadIdx.x;
  if (idx < 24 * 4 * 32 * 16) {                              // 49152: qkv_w 128x384
    const int i = idx & 15, lane = (idx >> 4) & 31, kk = (idx >> 9) & 3, jt = idx >> 11;
    const int k = kk * 32 + ((lane >> 4) << 4) + i;
    const int n = jt * 16 + (lane & 15);
    Wq[idx] = (bf16)qkv_w[k * 384 + n];
  } else if (idx < 24 * 4 * 32 * 16 + 8 * 4 * 32 * 16) {     // 16384: proj_w 128x128
    const int id2 = idx - 24 * 4 * 32 * 16;
    const int i = id2 & 15, lane = (id2 >> 4) & 31, kk = (id2 >> 9) & 3, jt = id2 >> 11;
    const int k = kk * 32 + ((lane >> 4) << 4) + i;
    const int n = jt * 16 + (lane & 15);
    Wp[id2] = (bf16)proj_w[k * 128 + n];
  }
}

// ---- kernel 1: QKV GEMM + scatter (Q row-major scaled, KT/V B-swizzled) ---
__global__ __launch_bounds__(256) void qkv_kernel(
    const float* __restrict__ x, const bf16* __restrict__ Wq,
    const float* __restrict__ qkv_b, bf16* __restrict__ Q,
    bf16* __restrict__ KTsw, bf16* __restrict__ Vsw) {
  __shared__ __align__(16) bf16 xs[2048];
  __shared__ uint32_t qb[16], kb2[16], vb[16];   // per-token scatter base offsets
  const int tid = threadIdx.x, lane = tid & 31, wave = tid >> 5;
  const int t0 = blockIdx.x * 16;

  // token -> (window, in-window) decomposition: once per block, 16 threads
  if (tid < 16) {
    const int t = t0 + tid;
    const int bi = t / 3136, p = t - bi * 3136;
    const int y = p / 56, xc = p - y * 56;
    const uint32_t win = ((bi << 3) + y / 7) * 8 + xc / 7;
    const uint32_t l = (y % 7) * 7 + (xc % 7);
    qb[tid]  = win * 8192 + l * 32;
    kb2[tid] = win * 8192 + ((l >> 4) << 9) + ((l & 15) << 4);
    vb[tid]  = win * 8192 + ((l >> 5) << 9) + (((l >> 4) & 1) << 8) + (l & 15);
  }
#if HAVE_TDM
  __shared__ __align__(16) float xf[2048];
  if (wave == 0) { tdm_copy_1d(xf, x + (size_t)t0 * 128, 2048, 2u); tdm_wait(); }
  __syncthreads();
  for (int i = tid; i < 2048; i += 256) xs[i] = (bf16)xf[i];
#else
  for (int i = tid; i < 2048; i += 256) xs[i] = (bf16)x[(size_t)t0 * 128 + i];
#endif
  __syncthreads();

  const int m = lane & 15, kb = (lane >> 4) << 4;
  for (int jt = wave; jt < 24; jt += 8) {
    v8f acc = {};
#pragma unroll
    for (int kk = 0; kk < 4; ++kk) {
      v16bf a = load_A16(xs + m * 128 + kk * 32 + kb);
      v16bf b = *(const v16bf*)(Wq + (((jt << 2) + kk) * 32 + lane) * 16);
      acc = wmma_bf16(a, b, acc);
    }
    const int n = jt * 16 + (lane & 15);
    const int which = n >> 7, rem = n & 127, head = rem >> 5, d = rem & 31;
    const float bias = qkv_b[n];
    // per-lane constant part of the scatter offset (uniform over r)
    uint32_t add;
    if (which == 0)      add = head * 2048 + d;
    else if (which == 1) add = head * 2048 + ((d >> 4) << 8) + (d & 15);
    else                 add = head * 2048 + ((d >> 4) << 10) + ((d & 15) << 4);
#pragma unroll
    for (int r = 0; r < 8; ++r) {
      const int row = r + ((lane >> 4) << 3);
      const float v = acc[r] + bias;
      if (which == 0)      Q[qb[row] + add]    = (bf16)(v * SCALE);
      else if (which == 1) KTsw[kb2[row] + add] = (bf16)v;
      else                 Vsw[vb[row] + add]   = (bf16)v;
    }
  }
}

// ---- kernel 2: per-(window,head) attention --------------------------------
__global__ __launch_bounds__(128) void attn_kernel(
    const bf16* __restrict__ Q, const bf16* __restrict__ KTsw,
    const bf16* __restrict__ Vsw, const int* __restrict__ rel_idx,
    const float* __restrict__ rel_table, const float* __restrict__ amask,
    bf16* __restrict__ Y) {
  __shared__ __align__(16) bf16 qs[2048];
  __shared__ __align__(16) bf16 kts[2048];
  __shared__ __align__(16) bf16 vs[2048];
  __shared__ __align__(16) bf16 attns[LP * LP];
  __shared__ uint32_t yoff[LP];                  // token*128 or sentinel

  const int tid = threadIdx.x, lane = tid & 31, wave = tid >> 5;
  const int wh = blockIdx.x;
  const int win = wh >> 2, head = wh & 3;
  const int wi = win & 63;
  const size_t base = (size_t)wh * 2048;

  if (tid < LP) {
    if (tid < L_) {
      const int bi = win >> 6, wy = (win >> 3) & 7, wx = win & 7;
      const int y = wy * 7 + tid / 7, xc = wx * 7 + tid % 7;
      yoff[tid] = (uint32_t)(((bi * 56 + y) * 56 + xc) * 128);
    } else {
      yoff[tid] = 0xFFFFFFFFu;
    }
  }
#if HAVE_TDM
  if (wave == 0) {
    tdm_copy_1d(qs, Q + base, 2048, 1u);
    tdm_copy_1d(kts, KTsw + base, 2048, 1u);
    tdm_copy_1d(vs, Vsw + base, 2048, 1u);
    tdm_wait();
  }
#else
  {
    const uint32_t* Qg = (const uint32_t*)(Q + base);
    const uint32_t* Kg = (const uint32_t*)(KTsw + base);
    const uint32_t* Vg = (const uint32_t*)(Vsw + base);
    uint32_t* q32 = (uint32_t*)qs;
    uint32_t* k32 = (uint32_t*)kts;
    uint32_t* v32 = (uint32_t*)vs;
    for (int i = tid; i < 1024; i += 128) { q32[i] = Qg[i]; k32[i] = Kg[i]; v32[i] = Vg[i]; }
  }
#endif
  __syncthreads();

  // zero the 480 padded-V entries (tokens l>=49) so 0*pad stays exactly 0
  for (int z = tid; z < 480; z += 128) {
    const int dt = z / 240, r = z - dt * 240;
    const int ln = 16 + r / 15, i = 1 + r % 15;
    vs[(((dt << 1) + 1) * 32 + ln) * 16 + i] = (bf16)0.0f;
  }

  const int row0 = wave * 16;
  const int m = lane & 15, kb = (lane >> 4) << 4;

  // QK^T: K-dim = hd = 32 -> one WMMA per 16x16 logit tile
  v8f c[4];
  {
    v16bf a = load_A16(qs + (row0 + m) * HD + kb);
#pragma unroll
    for (int jt = 0; jt < 4; ++jt) {
      v16bf b = *(const v16bf*)(kts + (jt * 32 + lane) * 16);
      v8f z = {};
      c[jt] = wmma_bf16(a, b, z);
    }
  }

  // bias + mask + row softmax (rows live in 16-lane halves)
  const int jb = lane & 15;
#pragma unroll
  for (int r = 0; r < 8; ++r) {
    const int i = row0 + r + ((lane >> 4) << 3);
    float lg[4];
#pragma unroll
    for (int jt = 0; jt < 4; ++jt) {
      const int j = jt * 16 + jb;
      float v = c[jt][r];
      if (j >= L_)      v = -1e30f;
      else if (i >= L_) v = 0.0f;
      else v += rel_table[rel_idx[i * L_ + j] * HEADS + head] +
                amask[((size_t)wi * L_ + i) * L_ + j];
      lg[jt] = v;
    }
    float mx = fmaxf(fmaxf(lg[0], lg[1]), fmaxf(lg[2], lg[3]));
    for (int off = 8; off > 0; off >>= 1) mx = fmaxf(mx, __shfl_xor(mx, off, 32));
    float s = 0.0f;
#pragma unroll
    for (int jt = 0; jt < 4; ++jt) { lg[jt] = __expf(lg[jt] - mx); s += lg[jt]; }
    for (int off = 8; off > 0; off >>= 1) s += __shfl_xor(s, off, 32);
    const float inv = 1.0f / s;
#pragma unroll
    for (int jt = 0; jt < 4; ++jt)
      attns[i * LP + jt * 16 + jb] = (bf16)(lg[jt] * inv);
  }
  __syncthreads();   // attns complete + vs pads zeroed

  // AV: K-dim = 64 -> two WMMA per 16x16 tile, 2 d-tiles
#pragma unroll
  for (int dt = 0; dt < 2; ++dt) {
    v8f acc = {};
#pragma unroll
    for (int kk = 0; kk < 2; ++kk) {
      v16bf a = load_A16(attns + (row0 + m) * LP + kk * 32 + kb);
      v16bf b = *(const v16bf*)(vs + (((dt << 1) + kk) * 32 + lane) * 16);
      acc = wmma_bf16(a, b, acc);
    }
    const uint32_t dadd = head * 32 + dt * 16 + (lane & 15);
#pragma unroll
    for (int r = 0; r < 8; ++r) {
      const int i = row0 + r + ((lane >> 4) << 3);
      const uint32_t to = yoff[i];
      if (to != 0xFFFFFFFFu) Y[to + dadd] = (bf16)acc[r];
    }
  }
}

// ---- kernel 3: output projection, fp32 result -----------------------------
__global__ __launch_bounds__(256) void proj_kernel(
    const bf16* __restrict__ Y, const bf16* __restrict__ Wp,
    const float* __restrict__ proj_b, float* __restrict__ out) {
  __shared__ __align__(16) bf16 ys[2048];
  const int tid = threadIdx.x, lane = tid & 31, wave = tid >> 5;
  const size_t t0 = (size_t)blockIdx.x * 16;
#if HAVE_TDM
  if (wave == 0) { tdm_copy_1d(ys, Y + t0 * 128, 2048, 1u); tdm_wait(); }
#else
  {
    const uint32_t* Yg = (const uint32_t*)(Y + t0 * 128);
    uint32_t* y32 = (uint32_t*)ys;
    for (int i = tid; i < 1024; i += 256) y32[i] = Yg[i];
  }
#endif
  __syncthreads();

  const int jt = wave;                                      // 8 waves = 8 col tiles
  const int m = lane & 15, kb = (lane >> 4) << 4;
  v8f acc = {};
#pragma unroll
  for (int kk = 0; kk < 4; ++kk) {
    v16bf a = load_A16(ys + m * 128 + kk * 32 + kb);
    v16bf b = *(const v16bf*)(Wp + (((jt << 2) + kk) * 32 + lane) * 16);
    acc = wmma_bf16(a, b, acc);
  }
  const int n = jt * 16 + (lane & 15);
  const float bias = proj_b[n];
#pragma unroll
  for (int r = 0; r < 8; ++r) {
    const int row = r + ((lane >> 4) << 3);
    out[(t0 + row) * 128 + n] = acc[r] + bias;
  }
}

extern "C" void kernel_launch(void* const* d_in, const int* in_sizes, int n_in,
                              void* d_out, int out_size, void* d_ws, size_t ws_size,
                              hipStream_t stream) {
  (void)in_sizes; (void)n_in; (void)out_size; (void)ws_size;
  const float* x       = (const float*)d_in[0];
  const int*   rel_idx = (const int*)d_in[2];
  const float* amask   = (const float*)d_in[3];
  const float* qkv_w   = (const float*)d_in[4];
  const float* qkv_b   = (const float*)d_in[5];
  const float* rtable  = (const float*)d_in[6];
  const float* proj_w  = (const float*)d_in[7];
  const float* proj_b  = (const float*)d_in[8];
  float* out = (float*)d_out;

  // workspace (bf16): Q | KTsw | Vsw | Y | Wq | Wp   (~253 MB)
  const size_t per = (size_t)NWIN * HEADS * 2048;           // 33,554,432 each
  bf16* Q    = (bf16*)d_ws;
  bf16* KTsw = Q + per;
  bf16* Vsw  = KTsw + per;
  bf16* Y    = Vsw + per;                                   // NTOK*128
  bf16* Wq   = Y + (size_t)NTOK * 128;                      // 49152
  bf16* Wp   = Wq + 49152;                                  // 16384

  init_weights_kernel<<<256, 256, 0, stream>>>(qkv_w, proj_w, Wq, Wp);
  qkv_kernel<<<NTOK / 16, 256, 0, stream>>>(x, Wq, qkv_b, Q, KTsw, Vsw);
  attn_kernel<<<NWIN * HEADS, 128, 0, stream>>>(Q, KTsw, Vsw, rel_idx, rtable, amask, Y);
  proj_kernel<<<NTOK / 16, 256, 0, stream>>>(Y, Wp, proj_b, out);
}
